// TransformerLayer_2662879724354
// MI455X (gfx1250) — matrix operand order repaired and checked by reading, not compiled, not run on hardware
//
#include <hip/hip_runtime.h>

typedef __attribute__((ext_vector_type(16))) _Float16 v16h;
typedef __attribute__((ext_vector_type(8)))  float    v8f;
typedef __attribute__((ext_vector_type(4)))  int      v4i;

// ---------------------------------------------------------------------------
// CDNA5 async global->LDS copy (ASYNCcnt path), with sync fallback
// ---------------------------------------------------------------------------
#if __has_builtin(__builtin_amdgcn_global_load_async_to_lds_b128)
#define ATHENA_ASYNC 1
#endif

__device__ __forceinline__ void cp16_to_lds(const _Float16* g, _Float16* l) {
#ifdef ATHENA_ASYNC
  __builtin_amdgcn_global_load_async_to_lds_b128(
      (__attribute__((address_space(1))) v4i*)(v4i*)(void*)g,
      (__attribute__((address_space(3))) v4i*)(v4i*)l, 0, 0);
#else
  *(uint4*)l = *(const uint4*)g;
#endif
}

__device__ __forceinline__ void wait_cp_async() {
#ifdef ATHENA_ASYNC
#if __has_builtin(__builtin_amdgcn_s_wait_asynccnt)
  __builtin_amdgcn_s_wait_asynccnt(0);
#else
  asm volatile("s_wait_asynccnt 0x0" ::: "memory");
#endif
#endif
}

// ---------------------------------------------------------------------------
// WMMA helpers
// ---------------------------------------------------------------------------
__device__ __forceinline__ v8f wmma_f16(v16h a, v16h b, v8f c) {
  // D = A(16x32 f16) * B(32x16 f16) + C(16x16 f32)
  return __builtin_amdgcn_wmma_f32_16x16x32_f16(false, a, false, b, (short)0, c,
                                                false, false);
}

__device__ __forceinline__ v8f vzero8() {
  v8f z;
#pragma unroll
  for (int i = 0; i < 8; i++) z[i] = 0.f;
  return z;
}

// A-matrix fragment (16x32 f16): lane r holds row r; lane's 16 elements are
// K = 16*half + [0..15] (contiguous, matching B).
__device__ __forceinline__ v16h load_a_frag(const _Float16* row) {
  const int half = (threadIdx.x >> 4) & 1;
  union { v16h h; uint4 u[2]; } f;
  f.u[0] = *(const uint4*)(row + 16 * half);
  f.u[1] = *(const uint4*)(row + 16 * half + 8);
  return f.h;
}

// B-matrix fragment (32x16 f16): lane r holds column r; lane's 16 elements are
// K = 16*half + [0..15] (contiguous).
__device__ __forceinline__ v16h load_b_frag(const _Float16* col) {
  const int half = (threadIdx.x >> 4) & 1;
  union { v16h h; uint4 u[2]; } f;
  f.u[0] = *(const uint4*)(col + 16 * half);
  f.u[1] = *(const uint4*)(col + 16 * half + 8);
  return f.h;
}

__device__ __forceinline__ float redmax16(float v) {
#pragma unroll
  for (int m = 1; m < 16; m <<= 1) v = fmaxf(v, __shfl_xor(v, m, 32));
  return v;
}
__device__ __forceinline__ float redsum16(float v) {
#pragma unroll
  for (int m = 1; m < 16; m <<= 1) v += __shfl_xor(v, m, 32);
  return v;
}

// ---------------------------------------------------------------------------
// fp32 -> fp16 conversion
// ---------------------------------------------------------------------------
__global__ void cvt_f32_f16(const float* __restrict__ in,
                            _Float16* __restrict__ out, size_t n) {
  size_t i = (size_t)blockIdx.x * blockDim.x + threadIdx.x;
  size_t stride = (size_t)gridDim.x * blockDim.x;
  for (; i < n; i += stride) out[i] = (_Float16)in[i];
}

// ---------------------------------------------------------------------------
// Generic WMMA GEMM:  out[N,O] = A[N,K](f16) * W[O,K](f16)^T + bias
//                      (+ res[N,O] f32) (gelu) -> f16 and/or f32
// Block tile 128x128, 8 waves (2 in M x 4 in O), K step 32,
// double-buffered LDS with async global->LDS staging (1 barrier / K-step).
// ---------------------------------------------------------------------------
__device__ __forceinline__ void stage_tiles(const _Float16* __restrict__ A,
                                            const _Float16* __restrict__ W,
                                            _Float16* at, _Float16* wt,
                                            int mBase, int oBase, int N, int K,
                                            int kb, int tid) {
#pragma unroll
  for (int i = 0; i < 2; i++) {
    int c = i * 256 + tid;  // 512 chunks of 8 halfs per tile
    int row = c >> 2;
    int col = (c & 3) * 8;
    int gr = mBase + row;
    if (gr >= N) gr = N - 1;  // clamp: garbage rows are never stored
    int go = oBase + row;     // O is a multiple of 128
    cp16_to_lds(A + (size_t)gr * K + kb + col, &at[row * 40 + col]);
    cp16_to_lds(W + (size_t)go * K + kb + col, &wt[row * 40 + col]);
  }
}

template <bool RES, bool GELU, bool O16, bool O32>
__global__ __launch_bounds__(256) void gemm_wmma(
    const _Float16* __restrict__ A, const _Float16* __restrict__ W,
    const float* __restrict__ bias, const float* __restrict__ res,
    _Float16* __restrict__ out16, float* __restrict__ out32,
    int N, int K, int O) {
  __shared__ __align__(16) _Float16 At[2][128 * 40];  // 32-wide K slab, pad 8
  __shared__ __align__(16) _Float16 Wt[2][128 * 40];

  const int tid  = threadIdx.x;
  const int wid  = tid >> 5;
  const int lane = tid & 31;
  const int r    = lane & 15;
  const int half = lane >> 4;
  const int wm   = (wid & 1) * 64;   // wave M offset in block tile
  const int wn   = (wid >> 1) * 32;  // wave O offset in block tile
  const int mBase = blockIdx.y * 128;
  const int oBase = blockIdx.x * 128;

  v8f acc[4][2];
#pragma unroll
  for (int mi = 0; mi < 4; mi++)
#pragma unroll
    for (int oi = 0; oi < 2; oi++) acc[mi][oi] = vzero8();

  stage_tiles(A, W, At[0], Wt[0], mBase, oBase, N, K, 0, tid);
  int cur = 0;
  for (int kb = 0; kb < K; kb += 32) {
    wait_cp_async();      // own-wave async copies done
    __syncthreads();      // all waves' copies visible; prev compute finished
    if (kb + 32 < K)
      stage_tiles(A, W, At[cur ^ 1], Wt[cur ^ 1], mBase, oBase, N, K, kb + 32,
                  tid);
    const _Float16* at = At[cur];
    const _Float16* wt = Wt[cur];
    v16h bf0 = load_b_frag(&wt[(wn + r) * 40]);
    v16h bf1 = load_b_frag(&wt[(wn + 16 + r) * 40]);
    v16h af[4];
#pragma unroll
    for (int mi = 0; mi < 4; mi++)
      af[mi] = load_a_frag(&at[(wm + mi * 16 + r) * 40]);
#pragma unroll
    for (int mi = 0; mi < 4; mi++) {
      acc[mi][0] = wmma_f16(af[mi], bf0, acc[mi][0]);
      acc[mi][1] = wmma_f16(af[mi], bf1, acc[mi][1]);
    }
    cur ^= 1;
  }

  // epilogue: C layout -> lane holds (row = i + 8*half, col = r) per VGPR i
#pragma unroll
  for (int oi = 0; oi < 2; oi++) {
    int colg = oBase + wn + oi * 16 + r;
    float bb = bias[colg];
#pragma unroll
    for (int mi = 0; mi < 4; mi++) {
#pragma unroll
      for (int i = 0; i < 8; i++) {
        int rowg = mBase + wm + mi * 16 + 8 * half + i;
        if (rowg < N) {
          float vv = acc[mi][oi][i] + bb;
          if (RES) vv += res[(size_t)rowg * O + colg];
          if (GELU) vv = 0.5f * vv * (1.f + erff(vv * 0.70710678118654752f));
          if (O32) out32[(size_t)rowg * O + colg] = vv;
          if (O16) out16[(size_t)rowg * O + colg] = (_Float16)vv;
        }
      }
    }
  }
}

// ---------------------------------------------------------------------------
// Ragged flash attention. grid = (heads=16, segments). block = 256 (8 waves).
// qk: [N,2048] f16 (q = cols 0..1023, k = cols 1024..2047), v: [N,1024] f16.
// out: [N,1024] f16. d = 64, scale = 0.125.
// ---------------------------------------------------------------------------
__global__ __launch_bounds__(256) void attn_wmma(
    const _Float16* __restrict__ qk, const _Float16* __restrict__ vv,
    const int* __restrict__ cumsum, _Float16* __restrict__ out) {
  const int h = blockIdx.x;
  const int b = blockIdx.y;
  const int s0 = cumsum[b];
  const int L  = cumsum[b + 1] - s0;

  const int tid  = threadIdx.x;
  const int wid  = tid >> 5;
  const int lane = tid & 31;
  const int r    = lane & 15;
  const int half = lane >> 4;

  __shared__ __align__(16) _Float16 Kl[32 * 72];      // 32 keys x 64 d (pad 8)
  __shared__ __align__(16) _Float16 Vt[64 * 40];      // transposed: d x key
  __shared__ __align__(16) _Float16 Pl[8 * 16 * 40];  // per-wave P (16x32)
  _Float16* Pw = &Pl[wid * 16 * 40];

  const int numQt  = (L + 15) >> 4;
  const int numKt  = (L + 31) >> 5;
  const int rounds = (numQt + 7) >> 3;

  for (int rd = 0; rd < rounds; rd++) {
    const int qt = rd * 8 + wid;
    // Q fragments (A operand): lane r = query row; clamp to stay in segment.
    int ql = qt * 16 + r;
    if (ql >= L) ql = L - 1;
    const _Float16* qp = qk + (size_t)(s0 + ql) * 2048 + h * 64;
    v16h aq0 = load_a_frag(qp);       // d 0..31
    v16h aq1 = load_a_frag(qp + 32);  // d 32..63

    float m_i[8], l_i[8];
    v8f acc[4];
#pragma unroll
    for (int i = 0; i < 8; i++) { m_i[i] = -1e30f; l_i[i] = 0.f; }
#pragma unroll
    for (int t = 0; t < 4; t++) acc[t] = vzero8();

    for (int kt = 0; kt < numKt; kt++) {
      __syncthreads();  // protect Kl/Vt from previous iteration readers
      {                 // cooperative staging of 32 keys
        int row = tid >> 3;            // key in chunk
        int col = (tid & 7) * 8;       // d offset
        int kl  = kt * 32 + row;
        const bool valid = kl < L;
        size_t gk = (size_t)(s0 + (valid ? kl : L - 1));
        // K chunk: async copy; invalid rows carry garbage but their score
        // columns are overwritten with -1e30 before use.
        cp16_to_lds(qk + gk * 2048 + 1024 + h * 64 + col, &Kl[row * 72 + col]);
        // V chunk: must be zero-filled when invalid (0 * NaN = NaN), and is
        // transposed through VGPRs so the P*V B-fragments are contiguous.
        uint4 dv = make_uint4(0u, 0u, 0u, 0u);
        if (valid) dv = *(const uint4*)(vv + gk * 1024 + h * 64 + col);
        union { uint4 u; _Float16 e[8]; } vu; vu.u = dv;
#pragma unroll
        for (int j = 0; j < 8; j++) Vt[(col + j) * 40 + row] = vu.e[j];
      }
      wait_cp_async();
      __syncthreads();

      // S = Q * K^T for two 16-key subtiles (K-dim = 64 -> 2 wmma each)
      v8f sA = vzero8(), sB = vzero8();
      {
        v16h b0 = load_b_frag(&Kl[r * 72]);
        v16h b1 = load_b_frag(&Kl[r * 72 + 32]);
        v16h c0 = load_b_frag(&Kl[(16 + r) * 72]);
        v16h c1 = load_b_frag(&Kl[(16 + r) * 72 + 32]);
        sA = wmma_f16(aq0, b0, sA);
        sA = wmma_f16(aq1, b1, sA);
        sB = wmma_f16(aq0, c0, sB);
        sB = wmma_f16(aq1, c1, sB);
      }
      const int k0 = kt * 32 + r;
      const int k1 = kt * 32 + 16 + r;
#pragma unroll
      for (int i = 0; i < 8; i++) {
        float x0 = sA[i] * 0.125f; if (k0 >= L) x0 = -1e30f;
        float x1 = sB[i] * 0.125f; if (k1 >= L) x1 = -1e30f;
        float tm = redmax16(fmaxf(x0, x1));
        float mn = fmaxf(m_i[i], tm);
        float sc = __expf(m_i[i] - mn);
        float p0 = __expf(x0 - mn);
        float p1 = __expf(x1 - mn);
        float rs = redsum16(p0 + p1);
        l_i[i] = l_i[i] * sc + rs;
        m_i[i] = mn;
#pragma unroll
        for (int t = 0; t < 4; t++) acc[t][i] *= sc;
        // C layout (row = i + 8*half, col = r / 16+r) -> per-wave P buffer
        Pw[(i + 8 * half) * 40 + r]      = (_Float16)p0;
        Pw[(i + 8 * half) * 40 + 16 + r] = (_Float16)p1;
      }
      __syncthreads();  // P visible; also fences Kl reads vs next staging

      // O += P(16x32) * V(32x64): 4 output d-tiles
      v16h pa = load_a_frag(&Pw[r * 40]);
#pragma unroll
      for (int t = 0; t < 4; t++) {
        v16h bv = load_b_frag(&Vt[(t * 16 + r) * 40]);
        acc[t] = wmma_f16(pa, bv, acc[t]);
      }
    }

    // normalize + store valid rows
#pragma unroll
    for (int i = 0; i < 8; i++) {
      int qr = qt * 16 + i + 8 * half;
      if (qr < L) {
        float inv = 1.f / l_i[i];
        size_t o = (size_t)(s0 + qr) * 1024 + h * 64;
#pragma unroll
        for (int t = 0; t < 4; t++)
          out[o + t * 16 + r] = (_Float16)(acc[t][i] * inv);
      }
    }
  }
}

// ---------------------------------------------------------------------------
// LayerNorm over H=1024. One block per row. Writes f32 (+ optional f16 copy).
// ---------------------------------------------------------------------------
template <bool ALSO16>
__global__ __launch_bounds__(256) void ln_kernel(
    const float* __restrict__ x, const float* __restrict__ g,
    const float* __restrict__ bta, float* __restrict__ o32,
    _Float16* __restrict__ o16, int H) {
  const size_t row = blockIdx.x;
  const float* xr = x + row * H;
  float s = 0.f, s2 = 0.f;
  for (int c = threadIdx.x; c < H; c += 256) {
    float v = xr[c];
    s += v; s2 += v * v;
  }
  const int wid = threadIdx.x >> 5, lane = threadIdx.x & 31;
#pragma unroll
  for (int m = 1; m < 32; m <<= 1) {
    s  += __shfl_xor(s, m, 32);
    s2 += __shfl_xor(s2, m, 32);
  }
  __shared__ float aS[8], aS2[8];
  if (lane == 0) { aS[wid] = s; aS2[wid] = s2; }
  __syncthreads();
  if (threadIdx.x == 0) {
    float t = 0.f, t2 = 0.f;
    for (int i = 0; i < 8; i++) { t += aS[i]; t2 += aS2[i]; }
    aS[0] = t; aS2[0] = t2;
  }
  __syncthreads();
  const float mean = aS[0] / H;
  const float var  = aS2[0] / H - mean * mean;
  const float rstd = rsqrtf(var + 1e-5f);
  for (int c = threadIdx.x; c < H; c += 256) {
    float v = (xr[c] - mean) * rstd * g[c] + bta[c];
    o32[row * H + c] = v;
    if (ALSO16) o16[row * H + c] = (_Float16)v;
  }
}

// ---------------------------------------------------------------------------
// host launcher
// ---------------------------------------------------------------------------
extern "C" void kernel_launch(void* const* d_in, const int* in_sizes, int n_in,
                              void* d_out, int out_size, void* d_ws,
                              size_t ws_size, hipStream_t stream) {
  (void)n_in; (void)out_size; (void)ws_size;
  const int H = 1024;
  const int N = in_sizes[0] / H;
  const int B = in_sizes[15] - 1;

  const float* h_n    = (const float*)d_in[0];
  const float* W_v    = (const float*)d_in[1];
  const float* b_v    = (const float*)d_in[2];
  const float* W_qk   = (const float*)d_in[3];
  const float* b_qk   = (const float*)d_in[4];
  const float* W_proj = (const float*)d_in[5];
  const float* b_proj = (const float*)d_in[6];
  const float* W_ff1  = (const float*)d_in[7];
  const float* b_ff1  = (const float*)d_in[8];
  const float* W_ff2  = (const float*)d_in[9];
  const float* b_ff2  = (const float*)d_in[10];
  const float* ln1_g  = (const float*)d_in[11];
  const float* ln1_b  = (const float*)d_in[12];
  const float* ln2_g  = (const float*)d_in[13];
  const float* ln2_b  = (const float*)d_in[14];
  const int*   cumsum = (const int*)d_in[15];
  float* outF = (float*)d_out;

  char* ws = (char*)d_ws;
  size_t off = 0;
  auto alloc = [&](size_t bytes) -> char* {
    off = (off + 255) & ~(size_t)255;
    char* p = ws + off;
    off += bytes;
    return p;
  };
  const size_t NH = (size_t)N * H;
  _Float16* X16   = (_Float16*)alloc(2 * NH);
  _Float16* Wv16  = (_Float16*)alloc(2 * (size_t)H * H);
  _Float16* Wqk16 = (_Float16*)alloc(2 * (size_t)2 * H * H);
  _Float16* Wp16  = (_Float16*)alloc(2 * (size_t)H * H);
  _Float16* Wf116 = (_Float16*)alloc(2 * (size_t)4 * H * H);
  _Float16* Wf216 = (_Float16*)alloc(2 * (size_t)4 * H * H);
  _Float16* V16   = (_Float16*)alloc(2 * NH);
  _Float16* QK16  = (_Float16*)alloc(2 * 2 * NH);
  _Float16* A16   = (_Float16*)alloc(2 * NH);
  float*    R32   = (float*)alloc(4 * NH);      // residual-sum scratch (reused)
  float*    H32   = (float*)alloc(4 * NH);      // LN1 output f32
  _Float16* H16   = (_Float16*)alloc(2 * NH);   // LN1 output f16
  _Float16* FF16  = (_Float16*)alloc(2 * 4 * NH);

  auto cvt = [&](const float* src, _Float16* dst, size_t n) {
    int grid = (int)((n + 1023) / 1024);
    if (grid > 4096) grid = 4096;
    cvt_f32_f16<<<grid, 256, 0, stream>>>(src, dst, n);
  };
  cvt(h_n, X16, NH);
  cvt(W_v, Wv16, (size_t)H * H);
  cvt(W_qk, Wqk16, (size_t)2 * H * H);
  cvt(W_proj, Wp16, (size_t)H * H);
  cvt(W_ff1, Wf116, (size_t)4 * H * H);
  cvt(W_ff2, Wf216, (size_t)4 * H * H);

  const dim3 blk(256);
  const int mBlocks = (N + 127) / 128;

  // v = h @ W_v^T + b_v                         (f16 out)
  gemm_wmma<false, false, true, false><<<dim3(H / 128, mBlocks), blk, 0, stream>>>(
      X16, Wv16, b_v, nullptr, V16, nullptr, N, H, H);
  // qk = h @ W_qk^T + b_qk                      (f16 out, q|k)
  gemm_wmma<false, false, true, false><<<dim3(2 * H / 128, mBlocks), blk, 0, stream>>>(
      X16, Wqk16, b_qk, nullptr, QK16, nullptr, N, H, 2 * H);
  // ragged multi-head attention
  attn_wmma<<<dim3(16, B), blk, 0, stream>>>(QK16, V16, cumsum, A16);
  // R = attn @ W_proj^T + b_proj + h_n          (f32 out)
  gemm_wmma<true, false, false, true><<<dim3(H / 128, mBlocks), blk, 0, stream>>>(
      A16, Wp16, b_proj, h_n, nullptr, R32, N, H, H);
  // LN1 -> H32 (f32) + H16 (f16)
  ln_kernel<true><<<N, blk, 0, stream>>>(R32, ln1_g, ln1_b, H32, H16, H);
  // FF1 = gelu(H16 @ W_ff1^T + b_ff1)           (f16 out)
  gemm_wmma<false, true, true, false><<<dim3(4 * H / 128, mBlocks), blk, 0, stream>>>(
      H16, Wf116, b_ff1, nullptr, FF16, nullptr, N, H, 4 * H);
  // R = FF16 @ W_ff2^T + b_ff2 + H32            (f32 out)
  gemm_wmma<true, false, false, true><<<dim3(H / 128, mBlocks), blk, 0, stream>>>(
      FF16, Wf216, b_ff2, H32, nullptr, R32, N, 4 * H, H);
  // LN2 -> final output (f32)
  ln_kernel<false><<<N, blk, 0, stream>>>(R32, ln2_g, ln2_b, outF, nullptr, H);
}
